// SimplifiedTopologicalFeatures_34402688041036
// MI455X (gfx1250) — compile-verified, matrix operand order checked by reading, not run on hardware
//
#include <hip/hip_runtime.h>
#include <hip/hip_bf16.h>

typedef __attribute__((ext_vector_type(16))) _Float16 v16h;
typedef __attribute__((ext_vector_type(8)))  float    v8f;

#define NCOL 10
#define FDIM 20     // 2*NCOL
#define HID  64
#define IMG  64
#define NPIX 4096   // IMG*IMG

// Carry-free 5-bit -> stride-6 bit spread:
//   (x * SPREAD_MUL) & SPREAD_SEL puts bit i of x at bit 6*i.
// All 25 partial-product positions i+5j are distinct, so the multiply never
// carries, and the mask keeps exactly the j==i terms.
#define SPREAD_MUL 0x00108421u   // 1 + 2^5 + 2^10 + 2^15 + 2^20
#define SPREAD_SEL 0x01041041u   // bits 0,6,12,18,24

// ---------------------------------------------------------------------------
// Kernel 0: pad/convert weights to f16 WMMA-friendly layouts.
//   W1 (64,20) f32 -> W1h (64,32) f16, K padded with zeros (20..31)
//   W2 (20,64) f32 -> W2h (32,64) f16, N padded with zero rows (20..31)
// ---------------------------------------------------------------------------
__global__ void prep_kernel(const float* __restrict__ W1, const float* __restrict__ W2,
                            _Float16* __restrict__ W1h, _Float16* __restrict__ W2h) {
  const int t = threadIdx.x; // 64 threads
  if (t < HID) {
    for (int k = 0; k < 32; ++k) {
      float v = (k < FDIM) ? W1[t * FDIM + k] : 0.0f;
      W1h[t * 32 + k] = (_Float16)v;
    }
  }
  if (t < 32) {
    if (t < FDIM) {
      for (int k = 0; k < HID; ++k) W2h[t * HID + k] = (_Float16)W2[t * HID + k];
    } else {
      for (int k = 0; k < HID; ++k) W2h[t * HID + k] = (_Float16)0.0f;
    }
  }
}

// ---------------------------------------------------------------------------
// Kernel 1: per-image color histogram + 3x3 "color present in neighborhood"
// (separable OR of 10-bit color bitmasks). One 256-thread block per image.
// Histograms use packed 6-bit-field accumulators (carry-free multiply spread)
// so the kernel stays HBM-bound instead of VALU-bound.
// ---------------------------------------------------------------------------
__global__ __launch_bounds__(256) void feat_kernel(const int* __restrict__ grid,
                                                   float* __restrict__ feats) {
  __shared__ unsigned int msk[NPIX];                  // 16 KB: per-pixel color bitmask
  __shared__ unsigned int hor[NPIX];                  // 16 KB: horizontal 3-OR
  __shared__ unsigned long long sCnt[NCOL];           // lo32 = count, hi32 = pooled

  const int t = threadIdx.x;
  const int b = blockIdx.x;
  if (t < NCOL) sCnt[t] = 0ull;

  // packed accumulators: five 6-bit fields each (max 16 per field)
  unsigned int cLo = 0, cHi = 0;   // colors 0-4 / 5-9, plain counts
  unsigned int pLo = 0, pHi = 0;   // colors 0-4 / 5-9, pooled-presence counts

  // ---- pass 1: coalesced load, build bitmasks, packed histogram ----
  const int4* g4 = (const int4*)(grid + (size_t)b * NPIX);
  __builtin_prefetch(g4 + t, 0, 0);   // global_prefetch_b8
#pragma unroll
  for (int j = 0; j < 4; ++j) {
    const int idx = t + 256 * j;      // int4 index, coalesced
    int4 v = g4[idx];
    unsigned int m0 = 1u << v.x, m1 = 1u << v.y, m2 = 1u << v.z, m3 = 1u << v.w;
    cLo += ((m0 & 31u) * SPREAD_MUL) & SPREAD_SEL;
    cHi += (((m0 >> 5) & 31u) * SPREAD_MUL) & SPREAD_SEL;
    cLo += ((m1 & 31u) * SPREAD_MUL) & SPREAD_SEL;
    cHi += (((m1 >> 5) & 31u) * SPREAD_MUL) & SPREAD_SEL;
    cLo += ((m2 & 31u) * SPREAD_MUL) & SPREAD_SEL;
    cHi += (((m2 >> 5) & 31u) * SPREAD_MUL) & SPREAD_SEL;
    cLo += ((m3 & 31u) * SPREAD_MUL) & SPREAD_SEL;
    cHi += (((m3 >> 5) & 31u) * SPREAD_MUL) & SPREAD_SEL;
    ((uint4*)msk)[idx] = make_uint4(m0, m1, m2, m3);
  }
  __syncthreads();

  // ---- pass 2: horizontal OR (row-wise, 16 cols per thread) ----
  {
    const int row = t >> 2;
    const int c0  = (t & 3) << 4;
    const unsigned int* mr = msk + row * IMG;
    uint4 qa = ((const uint4*)(mr + c0))[0];
    uint4 qb = ((const uint4*)(mr + c0))[1];
    uint4 qc = ((const uint4*)(mr + c0))[2];
    uint4 qd = ((const uint4*)(mr + c0))[3];
    unsigned int L = (c0 > 0)        ? mr[c0 - 1]  : 0u;
    unsigned int R = (c0 + 16 < IMG) ? mr[c0 + 16] : 0u;
    unsigned int v[16] = {qa.x, qa.y, qa.z, qa.w, qb.x, qb.y, qb.z, qb.w,
                          qc.x, qc.y, qc.z, qc.w, qd.x, qd.y, qd.z, qd.w};
    unsigned int h[16];
    h[0] = L | v[0] | v[1];
#pragma unroll
    for (int i = 1; i < 15; ++i) h[i] = v[i - 1] | v[i] | v[i + 1];
    h[15] = v[14] | v[15] | R;
    unsigned int* hr = hor + row * IMG + c0;
    ((uint4*)hr)[0] = make_uint4(h[0],  h[1],  h[2],  h[3]);
    ((uint4*)hr)[1] = make_uint4(h[4],  h[5],  h[6],  h[7]);
    ((uint4*)hr)[2] = make_uint4(h[8],  h[9],  h[10], h[11]);
    ((uint4*)hr)[3] = make_uint4(h[12], h[13], h[14], h[15]);
  }
  __syncthreads();

  // ---- pass 3: vertical OR + packed pooled-presence histogram ----
  {
    const int row = t >> 2;
    const int c0  = (t & 3) << 4;
#pragma unroll
    for (int q = 0; q < 4; ++q) {
      uint4 mid = ((const uint4*)(hor + row * IMG + c0))[q];
      uint4 up  = (row > 0)       ? ((const uint4*)(hor + (row - 1) * IMG + c0))[q]
                                  : make_uint4(0, 0, 0, 0);
      uint4 dn  = (row < IMG - 1) ? ((const uint4*)(hor + (row + 1) * IMG + c0))[q]
                                  : make_uint4(0, 0, 0, 0);
      unsigned int n0 = mid.x | up.x | dn.x;
      unsigned int n1 = mid.y | up.y | dn.y;
      unsigned int n2 = mid.z | up.z | dn.z;
      unsigned int n3 = mid.w | up.w | dn.w;
      pLo += ((n0 & 31u) * SPREAD_MUL) & SPREAD_SEL;
      pHi += (((n0 >> 5) & 31u) * SPREAD_MUL) & SPREAD_SEL;
      pLo += ((n1 & 31u) * SPREAD_MUL) & SPREAD_SEL;
      pHi += (((n1 >> 5) & 31u) * SPREAD_MUL) & SPREAD_SEL;
      pLo += ((n2 & 31u) * SPREAD_MUL) & SPREAD_SEL;
      pHi += (((n2 >> 5) & 31u) * SPREAD_MUL) & SPREAD_SEL;
      pLo += ((n3 & 31u) * SPREAD_MUL) & SPREAD_SEL;
      pHi += (((n3 >> 5) & 31u) * SPREAD_MUL) & SPREAD_SEL;
    }
  }

  // ---- unpack packed fields + one ds_add_u64 per color ----
#pragma unroll
  for (int c = 0; c < 5; ++c) {
    const int sh = 6 * c;
    unsigned long long lo = (cLo >> sh) & 63u;           // count,  colors 0-4
    unsigned long long hi = (pLo >> sh) & 63u;           // pooled, colors 0-4
    atomicAdd(&sCnt[c], lo | (hi << 32));
    lo = (cHi >> sh) & 63u;                              // count,  colors 5-9
    hi = (pHi >> sh) & 63u;                              // pooled, colors 5-9
    atomicAdd(&sCnt[c + 5], lo | (hi << 32));
  }
  __syncthreads();
  if (t < 2 * NCOL) {
    const int c = t >> 1;
    const unsigned long long packed = sCnt[c];
    const unsigned int val = (t & 1) ? (unsigned int)(packed >> 32)   // pooled
                                     : (unsigned int)(packed & 0xffffffffu); // count
    feats[b * FDIM + t] = (float)val * (1.0f / (float)NPIX);
  }
}

// ---------------------------------------------------------------------------
// Kernel 2: MLP via WMMA. One wave32 per 16-row batch tile.
//   h   = relu(feats(16x20,pad32) x W1^T)  -> 4x v_wmma_f32_16x16x32_f16
//   out = h(16x64) x W2^T                  -> 4x v_wmma_f32_16x16x32_f16
// h is transposed into the A-layout through a wave-private LDS buffer
// (same-wave DS ordering; no barrier needed).
// ---------------------------------------------------------------------------
__global__ __launch_bounds__(256) void mlp_kernel(const float* __restrict__ feats,
                                                  const _Float16* __restrict__ W1h,
                                                  const float* __restrict__ b1,
                                                  const _Float16* __restrict__ W2h,
                                                  const float* __restrict__ b2,
                                                  float* __restrict__ out) {
  __shared__ __align__(32) _Float16 hshare[8 * 16 * HID]; // 16 KB (2 KB / wave)
  const int lane    = threadIdx.x & 31;
  const int wave    = threadIdx.x >> 5;
  const int rowbase = (blockIdx.x * 8 + wave) * 16;
  const int lm      = lane & 15;
  const bool hi     = lane >= 16;          // lanes 16..31 carry K=16..31
  const int kbase   = hi ? 16 : 0;

  // ---- A1: 16x32 f16 (K padded; only k<20 valid) straight from feats ----
  v16h a1 = {};
  {
    const float* fr = feats + (size_t)(rowbase + lm) * FDIM;
    if (!hi) {                         // K = 0..15 (all valid), 80B rows => 16B aligned
      const float4* f4 = (const float4*)fr;
      float4 q0 = f4[0], q1 = f4[1], q2 = f4[2], q3 = f4[3];
      a1[0] = (_Float16)q0.x;  a1[1] = (_Float16)q0.y;  a1[2] = (_Float16)q0.z;  a1[3] = (_Float16)q0.w;
      a1[4] = (_Float16)q1.x;  a1[5] = (_Float16)q1.y;  a1[6] = (_Float16)q1.z;  a1[7] = (_Float16)q1.w;
      a1[8] = (_Float16)q2.x;  a1[9] = (_Float16)q2.y;  a1[10] = (_Float16)q2.z; a1[11] = (_Float16)q2.w;
      a1[12] = (_Float16)q3.x; a1[13] = (_Float16)q3.y; a1[14] = (_Float16)q3.z; a1[15] = (_Float16)q3.w;
    } else {                           // K = 16..31; only 16..19 valid, rest stays 0
      float4 q0 = *(const float4*)(fr + 16);
      a1[0] = (_Float16)q0.x; a1[1] = (_Float16)q0.y; a1[2] = (_Float16)q0.z; a1[3] = (_Float16)q0.w;
    }
  }

  _Float16* hrow = hshare + wave * (16 * HID);
  const int mrow0 = hi ? 8 : 0;        // D layout: VGPR j holds M=j (lo lanes) / j+8 (hi lanes)

  // ---- layer 1: 4 N-tiles over HIDDEN=64, bias + relu, scatter to A-layout LDS ----
#pragma unroll
  for (int tN = 0; tN < 4; ++tN) {
    const int n = tN * 16 + lm;        // hidden unit == B-matrix column
    v16h bb = *(const v16h*)(W1h + n * 32 + kbase);
    v8f c = {};
    c = __builtin_amdgcn_wmma_f32_16x16x32_f16(false, a1, false, bb, (short)0, c, false, false);
    const float bias = b1[n];
#pragma unroll
    for (int j = 0; j < 8; ++j) {
      float hv = c[j] + bias;
      hv = hv > 0.0f ? hv : 0.0f;
      hrow[(mrow0 + j) * HID + n] = (_Float16)hv;
    }
  }

  // ---- layer 2: K=64 as 2 steps of 32; N=20 padded into 2 tiles of 16 ----
  v8f acc0 = {}, acc1 = {};
#pragma unroll
  for (int s = 0; s < 2; ++s) {
    v16h a2 = *(const v16h*)(hrow + lm * HID + s * 32 + kbase);  // A-layout read-back
    {
      const int n = lm;
      v16h bb = *(const v16h*)(W2h + n * HID + s * 32 + kbase);
      acc0 = __builtin_amdgcn_wmma_f32_16x16x32_f16(false, a2, false, bb, (short)0, acc0, false, false);
    }
    {
      const int n = 16 + lm;           // rows 20..31 of W2h are zero
      v16h bb = *(const v16h*)(W2h + n * HID + s * 32 + kbase);
      acc1 = __builtin_amdgcn_wmma_f32_16x16x32_f16(false, a2, false, bb, (short)0, acc1, false, false);
    }
  }

  // ---- epilogue: bias + store (only cols < 20) ----
  {
    const int col = lm;                // tile 0: cols 0..15, all valid
    const float bias = b2[col];
#pragma unroll
    for (int j = 0; j < 8; ++j)
      out[(size_t)(rowbase + mrow0 + j) * FDIM + col] = acc0[j] + bias;
  }
  {
    const int col = 16 + lm;           // tile 1: cols 16..31, valid < 20
    if (col < FDIM) {
      const float bias = b2[col];
#pragma unroll
      for (int j = 0; j < 8; ++j)
        out[(size_t)(rowbase + mrow0 + j) * FDIM + col] = acc1[j] + bias;
    }
  }
}

// ---------------------------------------------------------------------------
extern "C" void kernel_launch(void* const* d_in, const int* in_sizes, int n_in,
                              void* d_out, int out_size, void* d_ws, size_t ws_size,
                              hipStream_t stream) {
  const int*   grid = (const int*)d_in[0];
  const float* W1   = (const float*)d_in[1];
  const float* b1   = (const float*)d_in[2];
  const float* W2   = (const float*)d_in[3];
  const float* b2   = (const float*)d_in[4];
  float*       out  = (float*)d_out;

  const int B = in_sizes[0] / NPIX;    // 4096

  char* ws = (char*)d_ws;
  float*    feats = (float*)ws;                                   // B*20 f32
  size_t    off   = (size_t)B * FDIM * sizeof(float);             // 327,680 (32B aligned)
  _Float16* W1h   = (_Float16*)(ws + off);                        // 64*32 f16
  _Float16* W2h   = (_Float16*)(ws + off + HID * 32 * sizeof(_Float16));

  prep_kernel<<<1, 64, 0, stream>>>(W1, W2, W1h, W2h);
  feat_kernel<<<B, 256, 0, stream>>>(grid, feats);
  mlp_kernel<<<B / 128, 256, 0, stream>>>(feats, W1h, b1, W2h, b2, out);
}
